// GroupedQueryAttention_28501402977005
// MI455X (gfx1250) — matrix operand order repaired and checked by reading, not compiled, not run on hardware
//
#include <hip/hip_runtime.h>
#include <hip/hip_bf16.h>

typedef __attribute__((ext_vector_type(16))) _Float16 v16h;
typedef __attribute__((ext_vector_type(8)))  _Float16 v8h;
typedef __attribute__((ext_vector_type(8)))  float    v8f;

#define D_MODEL   2048
#define N_HEAD    32
#define N_KV_HEAD 8
#define HEAD_DIM  64
#define N_REP     4
#define BB        2
#define TT        2048

__device__ __forceinline__ v16h comb16(v8h lo, v8h hi) {
  return __builtin_shufflevector(lo, hi, 0,1,2,3,4,5,6,7,8,9,10,11,12,13,14,15);
}
__device__ __forceinline__ v8f wmma_f16(v16h a, v16h b, v8f c) {
  return __builtin_amdgcn_wmma_f32_16x16x32_f16(false, a, false, b, (short)0, c, false, false);
}

// ---------------- cast / layout kernels ----------------
__global__ void cast_f32_to_f16(const float* __restrict__ in, _Float16* __restrict__ out, int n) {
  int i = blockIdx.x * blockDim.x + threadIdx.x;
  if (i < n) out[i] = (_Float16)in[i];
}

// in: [K, N] f32 -> out: [N, K] f16 (so GEMM B-fragments read contiguous K)
__global__ void cast_transpose_f16(const float* __restrict__ in, _Float16* __restrict__ out,
                                   int K, int N) {
  int i = blockIdx.x * blockDim.x + threadIdx.x;
  if (i < K * N) {
    int k = i / N, n = i % N;
    out[(size_t)n * K + k] = (_Float16)in[i];
  }
}

// ---------------- WMMA GEMM: C[M,N] = A[M,K] * Bt[N,K]^T ----------------
// wave tile 32x64, block = 8 waves -> 256x64
template <typename OutT>
__global__ void gemm_wmma(const _Float16* __restrict__ A, const _Float16* __restrict__ Bt,
                          OutT* __restrict__ C, int M, int N, int K) {
  const int wave = threadIdx.x >> 5;
  const int lane = threadIdx.x & 31;
  const int half = lane >> 4, ln = lane & 15;
  const int n0 = blockIdx.x * 64;
  const int m0 = blockIdx.y * 256 + wave * 32;
  (void)M;

  v8f acc[2][4] = {};
  for (int k0 = 0; k0 < K; k0 += 32) {
    v16h a[2];
#pragma unroll
    for (int mi = 0; mi < 2; ++mi) {
      const _Float16* ap = A + (size_t)(m0 + mi * 16 + ln) * K + k0;
      v8h lo = *(const v8h*)(ap + half * 16);
      v8h hi = *(const v8h*)(ap + half * 16 + 8);
      a[mi] = comb16(lo, hi);
    }
#pragma unroll
    for (int nj = 0; nj < 4; ++nj) {
      const _Float16* bp = Bt + (size_t)(n0 + nj * 16 + ln) * K + k0 + half * 16;
      v8h lo = *(const v8h*)(bp);
      v8h hi = *(const v8h*)(bp + 8);
      v16h b = comb16(lo, hi);
#pragma unroll
      for (int mi = 0; mi < 2; ++mi)
        acc[mi][nj] = wmma_f16(a[mi], b, acc[mi][nj]);
    }
  }
#pragma unroll
  for (int mi = 0; mi < 2; ++mi)
#pragma unroll
    for (int nj = 0; nj < 4; ++nj)
#pragma unroll
      for (int v = 0; v < 8; ++v) {
        int m = m0 + mi * 16 + v + 8 * half;
        int n = n0 + nj * 16 + ln;
        C[(size_t)m * N + n] = (OutT)acc[mi][nj][v];
      }
}

// ---------------- RoPE + head-major relayout ----------------
// G: [B*T, nheads*64] f16 (gemm output)  ->  R: [B, nheads, T, 64] f16, rope applied
__global__ void rope_relayout(const _Float16* __restrict__ G, _Float16* __restrict__ R,
                              int nheads) {
  int idx = blockIdx.x * blockDim.x + threadIdx.x;
  int total = BB * TT * nheads * 32;
  if (idx >= total) return;
  int i = idx & 31;
  int tmp = idx >> 5;
  int h = tmp % nheads; tmp /= nheads;
  int t = tmp % TT;
  int b = tmp / TT;
  // inv_freq = 10000^(-2i/64) = exp(-i * ln(10000)/32)
  float inv_freq = __expf(-(float)i * (9.210340371976184f / 32.0f));
  float ang = (float)t * inv_freq;
  float s, c;
  __sincosf(ang, &s, &c);
  const _Float16* g = G + (size_t)(b * TT + t) * (nheads * 64) + h * 64;
  float x1 = (float)g[2 * i], x2 = (float)g[2 * i + 1];
  _Float16* r = R + ((size_t)(b * nheads + h) * TT + t) * 64;
  r[i]      = (_Float16)(x1 * c - x2 * s);
  r[32 + i] = (_Float16)(x1 * s + x2 * c);
}

// Vg: [B*T, 512] -> Vt: [B, 8, 64, T]   (so PV B-fragments read contiguous kv)
__global__ void transpose_v(const _Float16* __restrict__ Vg, _Float16* __restrict__ Vt) {
  int idx = blockIdx.x * blockDim.x + threadIdx.x;
  if (idx >= BB * TT * N_KV_HEAD * HEAD_DIM) return;
  int d = idx % 64; int tmp = idx / 64;
  int hk = tmp % N_KV_HEAD; tmp /= N_KV_HEAD;
  int t = tmp % TT; int b = tmp / TT;
  Vt[((size_t)(b * N_KV_HEAD + hk) * 64 + d) * TT + t] =
      Vg[(size_t)(b * TT + t) * (N_KV_HEAD * 64) + hk * 64 + d];
}

// ---------------- flash attention: 1 wave per 16-query tile ----------------
// Qr: [B,32,T,64]  Kr: [B,8,T,64]  Vt: [B,8,64,T]  attn: [B*T, 2048] f16
__global__ void flash_attn(const _Float16* __restrict__ Qr, const _Float16* __restrict__ Kr,
                           const _Float16* __restrict__ Vt, _Float16* __restrict__ attn,
                           const int* __restrict__ maskp) {
  __shared__ _Float16 plds[8 * 16 * 32];  // 1KB per wave, 8 waves
  const int wave = threadIdx.x >> 5;
  const int lane = threadIdx.x & 31;
  const int half = lane >> 4, ln = lane & 15;
  const int tiles = TT / 128;
  int t128 = blockIdx.x % tiles;
  int bh   = blockIdx.x / tiles;
  int h = bh % N_HEAD;
  int b = bh / N_HEAD;
  int hk = h / N_REP;
  int q0 = t128 * 128 + wave * 16;
  const int mask = maskp[0];

  // Q A-fragments (d 0..31 and 32..63)
  const _Float16* qb = Qr + ((size_t)(b * N_HEAD + h) * TT + q0 + ln) * 64;
  v16h a0 = comb16(*(const v8h*)(qb + half * 16), *(const v8h*)(qb + half * 16 + 8));
  v16h a1 = comb16(*(const v8h*)(qb + 32 + half * 16), *(const v8h*)(qb + 40 + half * 16));

  float mrun[8], lrun[8];
#pragma unroll
  for (int v = 0; v < 8; ++v) { mrun[v] = -__builtin_inff(); lrun[v] = 0.f; }
  v8f oacc[4] = {};

  const _Float16* kb_h = Kr + (size_t)(b * N_KV_HEAD + hk) * TT * 64;
  const _Float16* vb_h = Vt + (size_t)(b * N_KV_HEAD + hk) * 64 * TT;
  _Float16* lb = plds + wave * 16 * 32;

  const int kv_end = mask ? (q0 + 16) : TT;
  for (int s0 = 0; s0 < kv_end; s0 += 32) {
    // prefetch next kv step's K rows (global_prefetch_b8; K stream is L2-resident)
    if (s0 + 32 < kv_end) {
      __builtin_prefetch(kb_h + (size_t)(s0 + 32 + ln) * 64, 0, 1);
      __builtin_prefetch(kb_h + (size_t)(s0 + 48 + ln) * 64, 0, 1);
    }
    v8f st[2];
#pragma unroll
    for (int t2 = 0; t2 < 2; ++t2) {
      const _Float16* kb = kb_h + (size_t)(s0 + t2 * 16 + ln) * 64;
      v16h b0 = comb16(*(const v8h*)(kb + half * 16), *(const v8h*)(kb + half * 16 + 8));
      v16h b1 = comb16(*(const v8h*)(kb + 32 + half * 16), *(const v8h*)(kb + 40 + half * 16));
      v8f c = {};
      c = wmma_f16(a0, b0, c);
      c = wmma_f16(a1, b1, c);
      int col = s0 + t2 * 16 + ln;
#pragma unroll
      for (int v = 0; v < 8; ++v) {
        float sv = c[v] * 0.125f;                 // 1/sqrt(64)
        int row = q0 + v + 8 * half;
        if (mask && col > row) sv = -__builtin_inff();
        st[t2][v] = sv;
      }
    }
    // online softmax: row stats (row M=v+8*half lives across the 16 lanes of a half)
    float corr[8];
#pragma unroll
    for (int v = 0; v < 8; ++v) {
      float rm = fmaxf(st[0][v], st[1][v]);
      for (int off = 8; off >= 1; off >>= 1) rm = fmaxf(rm, __shfl_xor(rm, off, 32));
      float mn = fmaxf(mrun[v], rm);
      float p0 = __expf(st[0][v] - mn);
      float p1 = __expf(st[1][v] - mn);
      st[0][v] = p0; st[1][v] = p1;
      float rs = p0 + p1;
      for (int off = 8; off >= 1; off >>= 1) rs += __shfl_xor(rs, off, 32);
      corr[v] = __expf(mrun[v] - mn);
      lrun[v] = lrun[v] * corr[v] + rs;
      mrun[v] = mn;
    }
#pragma unroll
    for (int j = 0; j < 4; ++j)
#pragma unroll
      for (int v = 0; v < 8; ++v) oacc[j][v] = oacc[j][v] * corr[v];

    // P (C-layout) -> LDS row-major [16q x 32s] -> A-fragment
#pragma unroll
    for (int t2 = 0; t2 < 2; ++t2)
#pragma unroll
      for (int v = 0; v < 8; ++v)
        lb[(v + 8 * half) * 32 + t2 * 16 + ln] = (_Float16)st[t2][v];
    v16h pa = comb16(*(const v8h*)(lb + ln * 32 + half * 16),
                     *(const v8h*)(lb + ln * 32 + half * 16 + 8));

    // O += P @ V   (B-fragment from Vt: contiguous along kv; 4 independent accumulators)
#pragma unroll
    for (int j = 0; j < 4; ++j) {
      const _Float16* vb = vb_h + (size_t)(j * 16 + ln) * TT + s0 + half * 16;
      v16h bv = comb16(*(const v8h*)(vb), *(const v8h*)(vb + 8));
      oacc[j] = wmma_f16(pa, bv, oacc[j]);
    }
  }

  // normalize, store [b, q, h*64 + d]
#pragma unroll
  for (int v = 0; v < 8; ++v) {
    float inv = 1.0f / lrun[v];
    int q = q0 + v + 8 * half;
    _Float16* orow = attn + (size_t)(b * TT + q) * D_MODEL + h * 64;
#pragma unroll
    for (int j = 0; j < 4; ++j)
      orow[j * 16 + ln] = (_Float16)(oacc[j][v] * inv);
  }
}

// ---------------- host side ----------------
extern "C" void kernel_launch(void* const* d_in, const int* in_sizes, int n_in,
                              void* d_out, int out_size, void* d_ws, size_t ws_size,
                              hipStream_t stream) {
  (void)in_sizes; (void)n_in; (void)out_size; (void)ws_size;
  const float* x  = (const float*)d_in[0];
  const float* Wq = (const float*)d_in[1];
  const float* Wk = (const float*)d_in[2];
  const float* Wv = (const float*)d_in[3];
  const float* Wo = (const float*)d_in[4];
  const int* maskp = (const int*)d_in[5];
  float* out = (float*)d_out;

  char* ws = (char*)d_ws;
  const size_t MT = (size_t)BB * TT;                         // 4096 rows
  _Float16* xh   = (_Float16*)(ws + 0);                      // 16 MB
  _Float16* WqT  = (_Float16*)(ws + (16u << 20));            //  8 MB
  _Float16* WkT  = (_Float16*)(ws + (24u << 20));            //  2 MB
  _Float16* WvT  = (_Float16*)(ws + (26u << 20));            //  2 MB
  _Float16* WoT  = (_Float16*)(ws + (28u << 20));            //  8 MB
  _Float16* Qg   = (_Float16*)(ws + (36u << 20));            // 16 MB
  _Float16* Kg   = (_Float16*)(ws + (52u << 20));            //  4 MB
  _Float16* Vg   = (_Float16*)(ws + (56u << 20));            //  4 MB
  _Float16* Qr   = (_Float16*)(ws + (60u << 20));            // 16 MB
  _Float16* Kr   = (_Float16*)(ws + (76u << 20));            //  4 MB
  _Float16* Vt   = (_Float16*)(ws + (80u << 20));            //  4 MB -> 84 MB total
  _Float16* attnb = xh;  // xh is dead after the QKV GEMMs; alias it

  const int THR = 256;
  int nx = (int)(MT * D_MODEL);
  cast_f32_to_f16<<<(nx + THR - 1) / THR, THR, 0, stream>>>(x, xh, nx);
  cast_transpose_f16<<<(D_MODEL * D_MODEL + THR - 1) / THR, THR, 0, stream>>>(Wq, WqT, D_MODEL, D_MODEL);
  cast_transpose_f16<<<(D_MODEL * 512 + THR - 1) / THR, THR, 0, stream>>>(Wk, WkT, D_MODEL, 512);
  cast_transpose_f16<<<(D_MODEL * 512 + THR - 1) / THR, THR, 0, stream>>>(Wv, WvT, D_MODEL, 512);
  cast_transpose_f16<<<(D_MODEL * D_MODEL + THR - 1) / THR, THR, 0, stream>>>(Wo, WoT, D_MODEL, D_MODEL);

  dim3 gq(D_MODEL / 64, (int)(MT / 256));
  gemm_wmma<_Float16><<<gq, THR, 0, stream>>>(xh, WqT, Qg, (int)MT, D_MODEL, D_MODEL);
  dim3 gkv(512 / 64, (int)(MT / 256));
  gemm_wmma<_Float16><<<gkv, THR, 0, stream>>>(xh, WkT, Kg, (int)MT, 512, D_MODEL);
  gemm_wmma<_Float16><<<gkv, THR, 0, stream>>>(xh, WvT, Vg, (int)MT, 512, D_MODEL);

  int nq = BB * TT * N_HEAD * 32;
  rope_relayout<<<(nq + THR - 1) / THR, THR, 0, stream>>>(Qg, Qr, N_HEAD);
  int nk = BB * TT * N_KV_HEAD * 32;
  rope_relayout<<<(nk + THR - 1) / THR, THR, 0, stream>>>(Kg, Kr, N_KV_HEAD);
  int nv = BB * TT * N_KV_HEAD * HEAD_DIM;
  transpose_v<<<(nv + THR - 1) / THR, THR, 0, stream>>>(Vg, Vt);

  flash_attn<<<BB * N_HEAD * (TT / 128), THR, 0, stream>>>(Qr, Kr, Vt, attnb, maskp);

  dim3 go(D_MODEL / 64, (int)(MT / 256));
  gemm_wmma<float><<<go, THR, 0, stream>>>(attnb, WoT, out, (int)MT, D_MODEL, D_MODEL);
}
